// LSTMEmbedder_67465346285619
// MI455X (gfx1250) — matrix operand order repaired and checked by reading, not compiled, not run on hardware
//
#include <hip/hip_runtime.h>
#include <hip/hip_bf16.h>

typedef __attribute__((ext_vector_type(16))) _Float16 v16h;
typedef __attribute__((ext_vector_type(8)))  float    v8f;

#define T_STEPS 1024
#define BB 64
#define HH 256
#define HSTRIDE 264    // LDS h row stride in halves (mult of 8 -> 16B aligned)
#define GSTRIDE 1032   // LDS gate row stride in floats

union Frag16 { v16h v; float4 q[2]; };
union Half16 { _Float16 h[16]; float4 q[2]; };

// A-matrix 16x32 f16 fragment (M x K), row-major source with given half-stride.
// Lanes 0-15: row M=lane, K in {k0..k0+7, k0+16..k0+23};
// lanes 16-31: row M=lane-16, K in {k0+8..k0+15, k0+24..k0+31}.
__device__ __forceinline__ v16h load_a_frag(const _Float16* base, int stride,
                                            int m0, int k0, int lane) {
  const int row = m0 + (lane & 15);
  const int kb  = k0 + ((lane >> 4) << 3);
  const _Float16* p = base + (size_t)row * stride + kb;
  Frag16 f;
  f.q[0] = *reinterpret_cast<const float4*>(p);
  f.q[1] = *reinterpret_cast<const float4*>(p + 16);
  return f.v;
}

// B-matrix 32x16 f16 fragment (K x N). Source holds W row-major [N][K]
// (row n of W == column n of W^T). Lane: column n0+lane%16, K order matches A fragment.
__device__ __forceinline__ v16h load_b_frag(const _Float16* base, int stride,
                                            int n0, int k0, int lane) {
  const int row = n0 + (lane & 15);
  const int kb  = k0 + ((lane >> 4) << 3);
  const _Float16* p = base + (size_t)row * stride + kb;
  Frag16 f;
  f.q[0] = *reinterpret_cast<const float4*>(p);
  f.q[1] = *reinterpret_cast<const float4*>(p + 16);
  return f.v;
}

__device__ __forceinline__ float sigmoidf_(float x) {
  return 1.0f / (1.0f + __expf(-x));
}

__global__ void cvt_f16_kernel(const float* __restrict__ s, _Float16* __restrict__ d, int n) {
  int i = blockIdx.x * 256 + threadIdx.x;
  if (i < n) d[i] = (_Float16)s[i];
}

// ---------------- Layer 0: bidirectional LSTM over scalar input ----------------
// grid.x = direction, 1024 threads (32 waves), persistent over T.
__global__ __launch_bounds__(1024) void lstm_layer0_kernel(
    const float* __restrict__ timein,      // [B, T]
    const float* __restrict__ Wih0,        // [2, 4H, 1]
    const float* __restrict__ b0,          // [2, 4H]
    const _Float16* __restrict__ Whh0,     // [2, 4H, H] f16
    _Float16* __restrict__ y1)             // [T, B, 2H] f16
{
  const int dir  = blockIdx.x;
  const int tid  = threadIdx.x;
  const int lane = tid & 31;
  const int wave = tid >> 5;
  const int mt   = wave >> 3;              // 4 M-tiles
  const int ng   = wave & 7;               // 8 groups of 8 N-tiles

  __shared__ _Float16 hbuf[BB][HSTRIDE];   // recurrent hidden, f16
  __shared__ float    gbuf[BB][GSTRIDE];   // raw gates, f32 (i|f|g|o)
  __shared__ float    wihbuf[4 * HH];
  __shared__ float    bbuf[4 * HH];
  __shared__ float    tcol[BB];

  const _Float16* Wd = Whh0 + (size_t)dir * 4 * HH * HH;

  for (int i = tid; i < 4 * HH; i += 1024) {
    wihbuf[i] = Wih0[(size_t)dir * 4 * HH + i];
    bbuf[i]   = b0  [(size_t)dir * 4 * HH + i];
  }
  for (int i = tid; i < BB * HSTRIDE; i += 1024) (&hbuf[0][0])[i] = (_Float16)0.0f;

  // pointwise-phase ownership: 16 consecutive j of row b, c state in regs
  const int pb = tid >> 4;
  const int pj = (tid & 15) << 4;
  float creg[16];
#pragma unroll
  for (int jj = 0; jj < 16; ++jj) creg[jj] = 0.0f;
  __syncthreads();

  for (int s = 0; s < T_STEPS; ++s) {
    const int t = dir ? (T_STEPS - 1 - s) : s;
    if (tid < BB) tcol[tid] = timein[(size_t)tid * T_STEPS + t];

    // ---- matmul: g = h @ Whh^T, 2 chunks of 4 N-tiles per wave ----
    const int rofs = (lane >> 4) * 8;
#pragma unroll
    for (int ch = 0; ch < 2; ++ch) {
      v8f acc[4];
#pragma unroll
      for (int ni = 0; ni < 4; ++ni) acc[ni] = (v8f){0,0,0,0,0,0,0,0};
      const int nt0 = ng * 8 + ch * 4;
#pragma unroll 4
      for (int kk = 0; kk < 8; ++kk) {
        const int k0 = kk * 32;
        v16h a = load_a_frag(&hbuf[0][0], HSTRIDE, mt * 16, k0, lane);
#pragma unroll
        for (int ni = 0; ni < 4; ++ni) {
          v16h bf = load_b_frag(Wd, HH, (nt0 + ni) * 16, k0, lane);
          acc[ni] = __builtin_amdgcn_wmma_f32_16x16x32_f16(false, a, false, bf,
                        (short)0, acc[ni], false, false);
        }
      }
#pragma unroll
      for (int ni = 0; ni < 4; ++ni) {
        const int ncol = (nt0 + ni) * 16 + (lane & 15);
#pragma unroll
        for (int r = 0; r < 8; ++r)
          gbuf[mt * 16 + rofs + r][ncol] = acc[ni][r];
      }
    }
    __syncthreads();   // gbuf complete; h reads done; tcol ready

    // ---- pointwise gates ----
    {
      const float xt = tcol[pb];
      Half16 hv;
#pragma unroll
      for (int jj = 0; jj < 16; ++jj) {
        const int j = pj + jj;
        const float gi = gbuf[pb][j        ] + xt * wihbuf[j        ] + bbuf[j        ];
        const float gf = gbuf[pb][j +   HH ] + xt * wihbuf[j +   HH ] + bbuf[j +   HH ];
        const float gg = gbuf[pb][j + 2*HH ] + xt * wihbuf[j + 2*HH ] + bbuf[j + 2*HH ];
        const float go = gbuf[pb][j + 3*HH ] + xt * wihbuf[j + 3*HH ] + bbuf[j + 3*HH ];
        const float c = sigmoidf_(gf) * creg[jj] + sigmoidf_(gi) * tanhf(gg);
        creg[jj] = c;
        hv.h[jj] = (_Float16)(sigmoidf_(go) * tanhf(c));
      }
      *reinterpret_cast<float4*>(&hbuf[pb][pj])     = hv.q[0];
      *reinterpret_cast<float4*>(&hbuf[pb][pj + 8]) = hv.q[1];
      _Float16* yo = y1 + ((size_t)t * BB + pb) * (2 * HH) + dir * HH + pj;
      *reinterpret_cast<float4*>(yo)     = hv.q[0];
      *reinterpret_cast<float4*>(yo + 8) = hv.q[1];
    }
    __syncthreads();   // h(t) visible; gbuf reads done before next overwrite
  }
}

// ---------------- Layer 1: bidirectional LSTM, fused input GEMM ----------------
__global__ __launch_bounds__(1024) void lstm_layer1_kernel(
    const _Float16* __restrict__ y1,       // [T, B, 2H] f16
    const _Float16* __restrict__ Wih1,     // [2, 4H, 2H] f16
    const _Float16* __restrict__ Whh1,     // [2, 4H, H] f16
    const float* __restrict__ b1,          // [2, 4H]
    float* __restrict__ last)              // [B, 2H] f32
{
  const int dir  = blockIdx.x;
  const int tid  = threadIdx.x;
  const int lane = tid & 31;
  const int wave = tid >> 5;
  const int mt   = wave >> 3;
  const int ng   = wave & 7;

  __shared__ _Float16 hbuf[BB][HSTRIDE];
  __shared__ float    gbuf[BB][GSTRIDE];
  __shared__ float    bbuf[4 * HH];

  const _Float16* Wi = Wih1 + (size_t)dir * 4 * HH * (2 * HH);
  const _Float16* Wh = Whh1 + (size_t)dir * 4 * HH * HH;

  for (int i = tid; i < 4 * HH; i += 1024) bbuf[i] = b1[(size_t)dir * 4 * HH + i];
  for (int i = tid; i < BB * HSTRIDE; i += 1024) (&hbuf[0][0])[i] = (_Float16)0.0f;

  const int pb = tid >> 4;
  const int pj = (tid & 15) << 4;
  float creg[16];
#pragma unroll
  for (int jj = 0; jj < 16; ++jj) creg[jj] = 0.0f;
  __syncthreads();

  for (int s = 0; s < T_STEPS; ++s) {
    const int t = dir ? (T_STEPS - 1 - s) : s;
    const _Float16* ybase = y1 + (size_t)t * BB * (2 * HH);

    // warm next timestep's activations (global_prefetch_b8)
    {
      const int tn = dir ? (t > 0 ? t - 1 : 0) : (t < T_STEPS - 1 ? t + 1 : t);
      __builtin_prefetch(y1 + (size_t)tn * BB * (2 * HH) + (size_t)tid * 64, 0, 1);
    }

    const int rofs = (lane >> 4) * 8;
#pragma unroll
    for (int ch = 0; ch < 2; ++ch) {
      v8f acc[4];
#pragma unroll
      for (int ni = 0; ni < 4; ++ni) acc[ni] = (v8f){0,0,0,0,0,0,0,0};
      const int nt0 = ng * 8 + ch * 4;

      // input contribution: K = 512, A from global y1[t]
#pragma unroll 4
      for (int kk = 0; kk < 16; ++kk) {
        const int k0 = kk * 32;
        v16h a = load_a_frag(ybase, 2 * HH, mt * 16, k0, lane);
#pragma unroll
        for (int ni = 0; ni < 4; ++ni) {
          v16h bf = load_b_frag(Wi, 2 * HH, (nt0 + ni) * 16, k0, lane);
          acc[ni] = __builtin_amdgcn_wmma_f32_16x16x32_f16(false, a, false, bf,
                        (short)0, acc[ni], false, false);
        }
      }
      // recurrent contribution: K = 256, A from LDS h
#pragma unroll 4
      for (int kk = 0; kk < 8; ++kk) {
        const int k0 = kk * 32;
        v16h a = load_a_frag(&hbuf[0][0], HSTRIDE, mt * 16, k0, lane);
#pragma unroll
        for (int ni = 0; ni < 4; ++ni) {
          v16h bf = load_b_frag(Wh, HH, (nt0 + ni) * 16, k0, lane);
          acc[ni] = __builtin_amdgcn_wmma_f32_16x16x32_f16(false, a, false, bf,
                        (short)0, acc[ni], false, false);
        }
      }
#pragma unroll
      for (int ni = 0; ni < 4; ++ni) {
        const int ncol = (nt0 + ni) * 16 + (lane & 15);
#pragma unroll
        for (int r = 0; r < 8; ++r)
          gbuf[mt * 16 + rofs + r][ncol] = acc[ni][r];
      }
    }
    __syncthreads();

    {
      const bool lastStep = (s == T_STEPS - 1);
      Half16 hv;
#pragma unroll
      for (int jj = 0; jj < 16; ++jj) {
        const int j = pj + jj;
        const float gi = gbuf[pb][j        ] + bbuf[j        ];
        const float gf = gbuf[pb][j +   HH ] + bbuf[j +   HH ];
        const float gg = gbuf[pb][j + 2*HH ] + bbuf[j + 2*HH ];
        const float go = gbuf[pb][j + 3*HH ] + bbuf[j + 3*HH ];
        const float c = sigmoidf_(gf) * creg[jj] + sigmoidf_(gi) * tanhf(gg);
        creg[jj] = c;
        const float h = sigmoidf_(go) * tanhf(c);
        hv.h[jj] = (_Float16)h;
        if (lastStep) last[(size_t)pb * (2 * HH) + dir * HH + j] = h;
      }
      *reinterpret_cast<float4*>(&hbuf[pb][pj])     = hv.q[0];
      *reinterpret_cast<float4*>(&hbuf[pb][pj + 8]) = hv.q[1];
    }
    __syncthreads();
  }
}

// ---------------- Projection [64,512]x[512,256] + LayerNorm ----------------
__global__ __launch_bounds__(1024) void proj_ln_kernel(
    const float* __restrict__ last,        // [B, 2H] f32
    const _Float16* __restrict__ Wp,       // [E, 2H] f16
    const float* __restrict__ bproj,
    const float* __restrict__ gammav,
    const float* __restrict__ betav,
    float* __restrict__ out)               // [B, E]
{
  __shared__ _Float16 A16[BB][520];
  __shared__ float    ob[BB][257];

  const int tid = threadIdx.x, lane = tid & 31, wave = tid >> 5;

  for (int i = tid; i < BB * 2 * HH; i += 1024)
    A16[i >> 9][i & 511] = (_Float16)last[i];
  __syncthreads();

  const int mt  = wave & 3;
  const int ntb = (wave >> 2) * 2;
  v8f acc[2];
  acc[0] = (v8f){0,0,0,0,0,0,0,0};
  acc[1] = (v8f){0,0,0,0,0,0,0,0};

#pragma unroll 4
  for (int kk = 0; kk < 16; ++kk) {
    const int k0 = kk * 32;
    v16h a = load_a_frag(&A16[0][0], 520, mt * 16, k0, lane);
#pragma unroll
    for (int ni = 0; ni < 2; ++ni) {
      v16h bf = load_b_frag(Wp, 2 * HH, (ntb + ni) * 16, k0, lane);
      acc[ni] = __builtin_amdgcn_wmma_f32_16x16x32_f16(false, a, false, bf,
                    (short)0, acc[ni], false, false);
    }
  }

  const int rofs = (lane >> 4) * 8;
#pragma unroll
  for (int ni = 0; ni < 2; ++ni) {
    const int e = (ntb + ni) * 16 + (lane & 15);
    const float bp = bproj[e];
#pragma unroll
    for (int r = 0; r < 8; ++r)
      ob[mt * 16 + rofs + r][e] = acc[ni][r] + bp;
  }
  __syncthreads();

  if (tid < BB) {
    float mu = 0.0f;
    for (int j = 0; j < 256; ++j) mu += ob[tid][j];
    mu *= (1.0f / 256.0f);
    float var = 0.0f;
    for (int j = 0; j < 256; ++j) { float d = ob[tid][j] - mu; var += d * d; }
    const float inv = rsqrtf(var * (1.0f / 256.0f) + 1e-5f);
    for (int j = 0; j < 256; ++j)
      out[(size_t)tid * 256 + j] = gammav[j] * (ob[tid][j] - mu) * inv + betav[j];
  }
}

extern "C" void kernel_launch(void* const* d_in, const int* in_sizes, int n_in,
                              void* d_out, int out_size, void* d_ws, size_t ws_size,
                              hipStream_t stream) {
  const float* timein = (const float*)d_in[0];
  const float* Wih0   = (const float*)d_in[1];
  const float* Whh0   = (const float*)d_in[2];
  const float* b0     = (const float*)d_in[3];
  const float* Wih1   = (const float*)d_in[4];
  const float* Whh1   = (const float*)d_in[5];
  const float* b1     = (const float*)d_in[6];
  const float* Wproj  = (const float*)d_in[7];
  const float* bproj  = (const float*)d_in[8];
  const float* gammav = (const float*)d_in[9];
  const float* betav  = (const float*)d_in[10];

  _Float16* y1      = (_Float16*)d_ws;                        // 33,554,432 halves
  _Float16* whh0_16 = y1      + (size_t)T_STEPS * BB * 2 * HH;
  _Float16* wih1_16 = whh0_16 + (size_t)2 * 4 * HH * HH;
  _Float16* whh1_16 = wih1_16 + (size_t)2 * 4 * HH * 2 * HH;
  _Float16* wproj16 = whh1_16 + (size_t)2 * 4 * HH * HH;
  float*    lastbuf = (float*)(wproj16 + (size_t)HH * 2 * HH);

  const int n_whh0 = 2 * 4 * HH * HH;
  const int n_wih1 = 2 * 4 * HH * 2 * HH;
  const int n_whh1 = 2 * 4 * HH * HH;
  const int n_wp   = HH * 2 * HH;
  cvt_f16_kernel<<<(n_whh0 + 255) / 256, 256, 0, stream>>>(Whh0,  whh0_16, n_whh0);
  cvt_f16_kernel<<<(n_wih1 + 255) / 256, 256, 0, stream>>>(Wih1,  wih1_16, n_wih1);
  cvt_f16_kernel<<<(n_whh1 + 255) / 256, 256, 0, stream>>>(Whh1,  whh1_16, n_whh1);
  cvt_f16_kernel<<<(n_wp   + 255) / 256, 256, 0, stream>>>(Wproj, wproj16, n_wp);

  lstm_layer0_kernel<<<2, 1024, 0, stream>>>(timein, Wih0, b0, whh0_16, y1);
  lstm_layer1_kernel<<<2, 1024, 0, stream>>>(y1, wih1_16, whh1_16, b1, lastbuf);
  proj_ln_kernel<<<1, 1024, 0, stream>>>(lastbuf, wproj16, bproj, gammav, betav,
                                         (float*)d_out);
}
